// Decoder_67422396613099
// MI455X (gfx1250) — compile-verified, matrix operand order checked by reading, not run on hardware
//
#include <hip/hip_runtime.h>
#include <stdint.h>
#include <stddef.h>

#define B_    32
#define NCTX_ 512
#define EMB_  512
#define NMEL_ 80
#define T_    500
#define PRE_  256
#define ARNN_ 1024
#define DRNN_ 1024
#define ADIM_ 128
#define NF_   32
#define KS_   31
#define PAD_  15

typedef __attribute__((ext_vector_type(16))) __bf16 v16bf;
typedef __attribute__((ext_vector_type(8)))  float v8f;

__device__ __forceinline__ float sigf(float x) { return 1.0f / (1.0f + __expf(-x)); }

// ---------------------------------------------------------------------------
// Utility kernels
// ---------------------------------------------------------------------------
__global__ void f32_to_bf16_kernel(const float* __restrict__ in, __bf16* __restrict__ out, int n) {
  int i = blockIdx.x * blockDim.x + threadIdx.x;
  if (i < n) out[i] = (__bf16)in[i];
}

__global__ void zero_f32_kernel(float* p, int n) {
  int i = blockIdx.x * blockDim.x + threadIdx.x;
  if (i < n) p[i] = 0.0f;
}

__global__ void zero_bf16_kernel(__bf16* p, int n) {
  int i = blockIdx.x * blockDim.x + threadIdx.x;
  if (i < n) p[i] = (__bf16)0.0f;
}

// proc_mem = memory @ m_w^T   : (B*NCTX, EMB) x (ADIM, EMB) -> (B*NCTX, ADIM)
__global__ void proc_mem_kernel(const float* __restrict__ memory,
                                const float* __restrict__ m_w,
                                float* __restrict__ pm) {
  int idx = blockIdx.x * blockDim.x + threadIdx.x;   // row*ADIM + d
  int row = idx >> 7;
  int d   = idx & 127;
  const float* mrow = memory + (size_t)row * EMB_;
  const float* wrow = m_w + (size_t)d * EMB_;
  float acc = 0.0f;
  for (int k = 0; k < EMB_; ++k) acc += mrow[k] * wrow[k];
  pm[idx] = acc;
}

// prenet layer 1: h1[t,b,j] = relu( x[t,b,:80] @ w1[j,:80] ), x[0]=zeros,
// x[t>=1] = decoder_inputs[:, :, t-1]  (layout (B, 80, T))
__global__ void prenet1_kernel(const float* __restrict__ dec,
                               const float* __restrict__ w1,
                               float* __restrict__ h1) {
  int idx = blockIdx.x * blockDim.x + threadIdx.x;   // (t*B + b)*PRE + j
  int j   = idx & (PRE_ - 1);
  int row = idx >> 8;
  int b   = row & (B_ - 1);
  int t   = row >> 5;
  float acc = 0.0f;
  if (t > 0) {
    const float* x = dec + (size_t)b * NMEL_ * T_ + (t - 1);
    const float* w = w1 + (size_t)j * NMEL_;
    for (int m = 0; m < NMEL_; ++m) acc += x[(size_t)m * T_] * w[m];
  }
  h1[idx] = fmaxf(acc, 0.0f);
}

// prenet layer 2 -> bf16 activations for the per-step WMMA GEMMs
__global__ void prenet2_kernel(const float* __restrict__ h1,
                               const float* __restrict__ w2,
                               __bf16* __restrict__ xin) {
  int idx = blockIdx.x * blockDim.x + threadIdx.x;
  int j   = idx & (PRE_ - 1);
  int row = idx >> 8;
  const float* x = h1 + (size_t)row * PRE_;
  const float* w = w2 + (size_t)j * PRE_;
  float acc = 0.0f;
  for (int k = 0; k < PRE_; ++k) acc += x[k] * w[k];
  xin[idx] = (__bf16)fmaxf(acc, 0.0f);
}

// ---------------------------------------------------------------------------
// WMMA bf16 GEMM: z (32 x N, f32) = sum over segments of  x_seg @ w_seg^T
// x_seg: (32, K) bf16 row-major (row stride xs); w_seg: (N, K) bf16 (row stride ws)
// One wave computes a 32x16 output tile (2 M-tiles x 1 N-tile).
// A-frag layout (16-bit A 16x32): lane m=lane&15, lanes<16 hold K {0..7,16..23},
// lanes>=16 hold K {8..15,24..31} -> two contiguous 16B loads per fragment.
// ---------------------------------------------------------------------------
__device__ __forceinline__ void wmma_seg(v8f& acc0, v8f& acc1,
                                         const __bf16* __restrict__ x, int xs,
                                         const __bf16* __restrict__ w, int ws,
                                         int K, int ntile, int lane) {
  const int m  = lane & 15;
  const int kh = (lane >> 4) << 3;                 // 0 or 8
  const __bf16* xr0 = x + (size_t)m * xs;          // M-tile 0 row
  const __bf16* xr1 = x + (size_t)(16 + m) * xs;   // M-tile 1 row
  const __bf16* wr  = w + (size_t)(ntile * 16 + m) * ws;
  for (int k0 = 0; k0 < K; k0 += 32) {
    v16bf a0, a1, bf;
    ((uint4*)&a0)[0] = *(const uint4*)(xr0 + k0 + kh);
    ((uint4*)&a0)[1] = *(const uint4*)(xr0 + k0 + kh + 16);
    ((uint4*)&a1)[0] = *(const uint4*)(xr1 + k0 + kh);
    ((uint4*)&a1)[1] = *(const uint4*)(xr1 + k0 + kh + 16);
    ((uint4*)&bf)[0] = *(const uint4*)(wr  + k0 + kh);
    ((uint4*)&bf)[1] = *(const uint4*)(wr  + k0 + kh + 16);
    acc0 = __builtin_amdgcn_wmma_f32_16x16x32_bf16(false, a0, false, bf, (short)0, acc0, false, false);
    acc1 = __builtin_amdgcn_wmma_f32_16x16x32_bf16(false, a1, false, bf, (short)0, acc1, false, false);
  }
}

__global__ __launch_bounds__(256) void lstm_gemm_kernel(
    float* __restrict__ z, int N,
    const __bf16* __restrict__ x0, int xs0, const __bf16* __restrict__ w0, int ws0, int K0,
    const __bf16* __restrict__ x1, int xs1, const __bf16* __restrict__ w1, int ws1, int K1,
    const __bf16* __restrict__ x2, int xs2, const __bf16* __restrict__ w2, int ws2, int K2) {
  const int lane  = threadIdx.x & 31;
  const int wave  = threadIdx.x >> 5;
  const int ntile = blockIdx.x * (blockDim.x >> 5) + wave;   // wave-uniform
  if (ntile * 16 >= N) return;
  v8f acc0 = {}; v8f acc1 = {};
  wmma_seg(acc0, acc1, x0, xs0, w0, ws0, K0, ntile, lane);
  wmma_seg(acc0, acc1, x1, xs1, w1, ws1, K1, ntile, lane);
  wmma_seg(acc0, acc1, x2, xs2, w2, ws2, K2, ntile, lane);
  // C/D layout: VGPR r holds (M = r + 8*(lane>=16), N = lane&15)
  const int n  = ntile * 16 + (lane & 15);
  const int mb = (lane >> 4) * 8;
#pragma unroll
  for (int r = 0; r < 8; ++r) {
    z[(size_t)(mb + r) * N + n]      = acc0[r];
    z[(size_t)(16 + mb + r) * N + n] = acc1[r];
  }
}

// ---------------------------------------------------------------------------
// Fused: ARNN LSTM pointwise -> q -> location conv -> energies -> softmax -> context
// grid = B blocks, 512 threads (one thread per ctx position / per EMB dim)
// ---------------------------------------------------------------------------
__global__ __launch_bounds__(512) void attention_kernel(
    const float* __restrict__ z,
    const float* __restrict__ bih, const float* __restrict__ bhh,
    float* __restrict__ ac, __bf16* __restrict__ ah_bf,
    const float* __restrict__ q_w,
    float* __restrict__ aw_g, float* __restrict__ awc_g,
    const float* __restrict__ conv_w, const float* __restrict__ loc_w,
    const float* __restrict__ v_w, const float* __restrict__ proc_mem,
    const float* __restrict__ memory, const unsigned char* __restrict__ mask,
    float* __restrict__ actx, __bf16* __restrict__ actx_bf) {
  const int b = blockIdx.x;
  const int tid = threadIdx.x;
  __shared__ float s_aw[NCTX_], s_awc[NCTX_], s_ah[ARNN_], s_q[ADIM_], s_vw[ADIM_];
  __shared__ float s_cw[NF_ * 2 * KS_], s_lw[ADIM_ * NF_], s_e[NCTX_], s_tmp[NCTX_];

  s_aw[tid]  = aw_g[b * NCTX_ + tid];
  s_awc[tid] = awc_g[b * NCTX_ + tid];
  for (int i = tid; i < NF_ * 2 * KS_; i += 512) s_cw[i] = conv_w[i];
  for (int i = tid; i < ADIM_ * NF_; i += 512)   s_lw[i] = loc_w[i];
  if (tid < ADIM_) s_vw[tid] = v_w[tid];

  // ARNN LSTM pointwise (z already holds x@Wih^T + h@Whh^T)
  const float* zb = z + (size_t)b * 4 * ARNN_;
  for (int jj = tid; jj < ARNN_; jj += 512) {
    float iv = zb[jj]             + bih[jj]             + bhh[jj];
    float fv = zb[ARNN_ + jj]     + bih[ARNN_ + jj]     + bhh[ARNN_ + jj];
    float gv = zb[2 * ARNN_ + jj] + bih[2 * ARNN_ + jj] + bhh[2 * ARNN_ + jj];
    float ov = zb[3 * ARNN_ + jj] + bih[3 * ARNN_ + jj] + bhh[3 * ARNN_ + jj];
    float c2 = sigf(fv) * ac[b * ARNN_ + jj] + sigf(iv) * tanhf(gv);
    float h2 = sigf(ov) * tanhf(c2);
    ac[b * ARNN_ + jj] = c2;
    s_ah[jj] = h2;
    ah_bf[b * ARNN_ + jj] = (__bf16)h2;
  }
  __syncthreads();

  // q = ah @ q_w^T
  if (tid < ADIM_) {
    const float* w = q_w + (size_t)tid * ARNN_;
    float acc = 0.0f;
    for (int k = 0; k < ARNN_; ++k) acc += s_ah[k] * w[k];
    s_q[tid] = acc;
  }
  __syncthreads();

  // location conv (32 filters, 2 channels, ks=31) + energy for n = tid
  {
    const int n = tid;
    float loc_f[NF_];
#pragma unroll
    for (int f = 0; f < NF_; ++f) loc_f[f] = 0.0f;
    for (int tp = 0; tp < KS_; ++tp) {
      int ix = n - PAD_ + tp;
      if (ix >= 0 && ix < NCTX_) {
        float a0 = s_aw[ix], a1 = s_awc[ix];
#pragma unroll
        for (int f = 0; f < NF_; ++f)
          loc_f[f] += a0 * s_cw[f * (2 * KS_) + tp] + a1 * s_cw[f * (2 * KS_) + KS_ + tp];
      }
    }
    const float* pmr = proc_mem + ((size_t)b * NCTX_ + n) * ADIM_;
    float e_acc = 0.0f;
    for (int d = 0; d < ADIM_; ++d) {
      float l = 0.0f;
#pragma unroll
      for (int f = 0; f < NF_; ++f) l += loc_f[f] * s_lw[d * NF_ + f];
      e_acc += tanhf(s_q[d] + pmr[d] + l) * s_vw[d];
    }
    s_e[n] = mask[b * NCTX_ + n] ? -1.0e9f : e_acc;
  }
  __syncthreads();

  // softmax over NCTX
  s_tmp[tid] = s_e[tid];
  __syncthreads();
  for (int s = 256; s > 0; s >>= 1) {
    if (tid < s) s_tmp[tid] = fmaxf(s_tmp[tid], s_tmp[tid + s]);
    __syncthreads();
  }
  float mx = s_tmp[0];
  __syncthreads();
  float pexp = __expf(s_e[tid] - mx);
  s_tmp[tid] = pexp;
  __syncthreads();
  for (int s = 256; s > 0; s >>= 1) {
    if (tid < s) s_tmp[tid] += s_tmp[tid + s];
    __syncthreads();
  }
  float awn = pexp / s_tmp[0];
  __syncthreads();
  s_e[tid] = awn;                              // reuse s_e as attention weights
  aw_g[b * NCTX_ + tid]  = awn;
  awc_g[b * NCTX_ + tid] = s_awc[tid] + awn;
  __syncthreads();

  // context: actx[e] = sum_n aw[n] * memory[b, n, e]  (coalesced in e)
  {
    const float* mb = memory + (size_t)b * NCTX_ * EMB_;
    float acc = 0.0f;
    for (int n = 0; n < NCTX_; ++n) acc += s_e[n] * mb[(size_t)n * EMB_ + tid];
    actx[b * EMB_ + tid]    = acc;
    actx_bf[b * EMB_ + tid] = (__bf16)acc;
  }
}

// ---------------------------------------------------------------------------
// Fused: DRNN LSTM pointwise + mel/gate projection.  grid = B, 512 threads.
// ---------------------------------------------------------------------------
__global__ __launch_bounds__(512) void drnn_proj_kernel(
    const float* __restrict__ z,
    const float* __restrict__ bih, const float* __restrict__ bhh,
    float* __restrict__ dc, __bf16* __restrict__ dh_bf,
    const float* __restrict__ actx,
    const float* __restrict__ proj_w, const float* __restrict__ proj_b,
    const float* __restrict__ gate_w, const float* __restrict__ gate_b,
    float* __restrict__ out, int t) {
  const int b = blockIdx.x;
  const int tid = threadIdx.x;
  __shared__ float s_dhc[DRNN_ + EMB_];
  const float* zb = z + (size_t)b * 4 * DRNN_;
  for (int jj = tid; jj < DRNN_; jj += 512) {
    float iv = zb[jj]             + bih[jj]             + bhh[jj];
    float fv = zb[DRNN_ + jj]     + bih[DRNN_ + jj]     + bhh[DRNN_ + jj];
    float gv = zb[2 * DRNN_ + jj] + bih[2 * DRNN_ + jj] + bhh[2 * DRNN_ + jj];
    float ov = zb[3 * DRNN_ + jj] + bih[3 * DRNN_ + jj] + bhh[3 * DRNN_ + jj];
    float c2 = sigf(fv) * dc[b * DRNN_ + jj] + sigf(iv) * tanhf(gv);
    float h2 = sigf(ov) * tanhf(c2);
    dc[b * DRNN_ + jj] = c2;
    dh_bf[b * DRNN_ + jj] = (__bf16)h2;
    s_dhc[jj] = h2;
  }
  s_dhc[DRNN_ + tid] = actx[b * EMB_ + tid];   // tid covers all 512 EMB dims
  __syncthreads();
  if (tid < NMEL_) {
    const float* w = proj_w + (size_t)tid * (DRNN_ + EMB_);
    float acc = proj_b[tid];
    for (int k = 0; k < DRNN_ + EMB_; ++k) acc += s_dhc[k] * w[k];
    out[(size_t)b * NMEL_ * T_ + (size_t)tid * T_ + t] = acc;   // (B, 80, T)
  } else if (tid == NMEL_) {
    float acc = gate_b[0];
    for (int k = 0; k < DRNN_ + EMB_; ++k) acc += s_dhc[k] * gate_w[k];
    out[(size_t)B_ * NMEL_ * T_ + (size_t)b * T_ + t] = acc;    // (B, T)
  }
}

// ---------------------------------------------------------------------------
extern "C" void kernel_launch(void* const* d_in, const int* in_sizes, int n_in,
                              void* d_out, int out_size, void* d_ws, size_t ws_size,
                              hipStream_t stream) {
  (void)in_sizes; (void)n_in; (void)out_size; (void)ws_size;
  const float* memory    = (const float*)d_in[0];
  const float* dec       = (const float*)d_in[1];
  const unsigned char* mask = (const unsigned char*)d_in[2];
  const float* prenet_w1 = (const float*)d_in[3];
  const float* prenet_w2 = (const float*)d_in[4];
  const float* arnn_wih  = (const float*)d_in[5];
  const float* arnn_whh  = (const float*)d_in[6];
  const float* arnn_bih  = (const float*)d_in[7];
  const float* arnn_bhh  = (const float*)d_in[8];
  const float* q_w       = (const float*)d_in[9];
  const float* m_w       = (const float*)d_in[10];
  const float* conv_w    = (const float*)d_in[11];
  const float* loc_w     = (const float*)d_in[12];
  const float* v_w       = (const float*)d_in[13];
  const float* drnn_wih  = (const float*)d_in[14];
  const float* drnn_whh  = (const float*)d_in[15];
  const float* drnn_bih  = (const float*)d_in[16];
  const float* drnn_bhh  = (const float*)d_in[17];
  const float* proj_w    = (const float*)d_in[18];
  const float* proj_b    = (const float*)d_in[19];
  const float* gate_w    = (const float*)d_in[20];
  const float* gate_b    = (const float*)d_in[21];
  float* out = (float*)d_out;

  // carve workspace
  char* p = (char*)d_ws;
  auto alloc = [&](size_t bytes) -> char* {
    char* r = p;
    p += (bytes + 255) & ~(size_t)255;
    return r;
  };
  __bf16* wAih = (__bf16*)alloc((size_t)4096 * 768 * 2);
  __bf16* wAhh = (__bf16*)alloc((size_t)4096 * 1024 * 2);
  __bf16* wDih = (__bf16*)alloc((size_t)4096 * 1536 * 2);
  __bf16* wDhh = (__bf16*)alloc((size_t)4096 * 1024 * 2);
  __bf16* xin  = (__bf16*)alloc((size_t)T_ * B_ * PRE_ * 2);
  float*  pmem = (float*)alloc((size_t)B_ * NCTX_ * ADIM_ * 4);
  float*  h1   = (float*)alloc((size_t)T_ * B_ * PRE_ * 4);
  float*  z_a  = (float*)alloc((size_t)B_ * 4096 * 4);
  float*  z_d  = (float*)alloc((size_t)B_ * 4096 * 4);
  const int nf32 = B_ * ARNN_ + B_ * DRNN_ + B_ * NCTX_ * 2 + B_ * EMB_;
  float*  fst  = (float*)alloc((size_t)nf32 * 4);
  float*  ac   = fst;
  float*  dc   = ac + B_ * ARNN_;
  float*  aw   = dc + B_ * DRNN_;
  float*  awc  = aw + B_ * NCTX_;
  float*  actx = awc + B_ * NCTX_;
  const int nbf = B_ * ARNN_ + B_ * DRNN_ + B_ * EMB_;
  __bf16* bst  = (__bf16*)alloc((size_t)nbf * 2);
  __bf16* ah_bf   = bst;
  __bf16* dh_bf   = ah_bf + B_ * ARNN_;
  __bf16* actx_bf = dh_bf + B_ * DRNN_;

  // one-time (per call): bf16 weight conversion, proc_mem, prenet, state init
  auto cvt = [&](const float* s, __bf16* d, int n) {
    f32_to_bf16_kernel<<<(n + 255) / 256, 256, 0, stream>>>(s, d, n);
  };
  cvt(arnn_wih, wAih, 4096 * 768);
  cvt(arnn_whh, wAhh, 4096 * 1024);
  cvt(drnn_wih, wDih, 4096 * 1536);
  cvt(drnn_whh, wDhh, 4096 * 1024);
  proc_mem_kernel<<<(B_ * NCTX_ * ADIM_) / 256, 256, 0, stream>>>(memory, m_w, pmem);
  prenet1_kernel<<<(T_ * B_ * PRE_) / 256, 256, 0, stream>>>(dec, prenet_w1, h1);
  prenet2_kernel<<<(T_ * B_ * PRE_) / 256, 256, 0, stream>>>(h1, prenet_w2, xin);
  zero_f32_kernel<<<(nf32 + 255) / 256, 256, 0, stream>>>(fst, nf32);
  zero_bf16_kernel<<<(nbf + 255) / 256, 256, 0, stream>>>(bst, nbf);

  // sequential decode loop: stream order enforces the recurrence
  for (int t = 0; t < T_; ++t) {
    const __bf16* xt = xin + (size_t)t * B_ * PRE_;
    // z_a = [x | actx | h] @ [Wih | Whh]^T   (N=4096, 256 n-tiles, 8 waves/block)
    lstm_gemm_kernel<<<32, 256, 0, stream>>>(z_a, 4096,
        xt, PRE_,       wAih, 768, PRE_,
        actx_bf, EMB_,  wAih + PRE_, 768, EMB_,
        ah_bf, ARNN_,   wAhh, 1024, ARNN_);
    attention_kernel<<<B_, 512, 0, stream>>>(z_a, arnn_bih, arnn_bhh, ac, ah_bf,
        q_w, aw, awc, conv_w, loc_w, v_w, pmem, memory, mask, actx, actx_bf);
    lstm_gemm_kernel<<<32, 256, 0, stream>>>(z_d, 4096,
        ah_bf, ARNN_,   wDih, 1536, ARNN_,
        actx_bf, EMB_,  wDih + ARNN_, 1536, EMB_,
        dh_bf, DRNN_,   wDhh, 1024, DRNN_);
    drnn_proj_kernel<<<B_, 512, 0, stream>>>(z_d, drnn_bih, drnn_bhh, dc, dh_bf,
        actx, proj_w, proj_b, gate_w, gate_b, out, t);
  }
}
